// CGPBlock_90443421319292
// MI455X (gfx1250) — compile-verified, matrix operand order checked by reading, not run on hardware
//
#include <hip/hip_runtime.h>

#define BB 8
#define CC 128
#define LL 4096

typedef __attribute__((ext_vector_type(16))) _Float16 v16h;
typedef __attribute__((ext_vector_type(8)))  _Float16 v8h;
typedef __attribute__((ext_vector_type(8)))  float    v8f;
typedef __attribute__((ext_vector_type(4)))  unsigned int v4u;
typedef __attribute__((ext_vector_type(4)))  int      v4i;
typedef __attribute__((ext_vector_type(8)))  int      v8i;

// ---------------------------------------------------------------------------
// Fragment loader for 16-bit WMMA operands from a row-major f16 matrix.
// Per ISA layout (16-bit A 16x32 / B 32x16): lanes 0-15 hold K=0..7 (VGPR0-3)
// and K=16..23 (VGPR4-7); lanes 16-31 hold K=8..15 and K=24..31.
// ---------------------------------------------------------------------------
__device__ __forceinline__ v16h frag_rowmajor(const _Float16* base, int row,
                                              int stride, int col) {
  const _Float16* p = base + row * stride + col;
  v8h lo = *reinterpret_cast<const v8h*>(p);
  v8h hi = *reinterpret_cast<const v8h*>(p + 16);
  v16h r;
#pragma unroll
  for (int i = 0; i < 8; ++i) {
    r[i]     = lo[i];
    r[i + 8] = hi[i];
  }
  return r;
}

__device__ __forceinline__ v8f zero_v8f() {
  v8f z;
#pragma unroll
  for (int i = 0; i < 8; ++i) z[i] = 0.0f;
  return z;
}

// ---------------------------------------------------------------------------
// TDM: issue a 2D f16 tile load (tile_d1 rows x tile_d0 elems) into LDS with
// hardware row padding. Descriptor layout per CDNA5 ISA ch.8 (D# groups 0/1).
//   group0: [1:0]=count=1, [63:32]=lds byte addr, [120:64]=global addr,
//           [127:126]=type=2
//   group1: [17:16]=data_size(1->2B), [20]=pad_enable, [24:22]=pad_interval,
//           [31:25]=pad_amount, [79:48]=tensor_dim0, [111:80]=tensor_dim1,
//           [127:112]=tile_dim0, [143:128]=tile_dim1, [207:160]=dim0_stride
// ---------------------------------------------------------------------------
__device__ __forceinline__ void tdm_load_2d_f16(unsigned lds_byte_off,
                                                const _Float16* gaddr,
                                                unsigned tensor_d0,
                                                unsigned tensor_d1,
                                                unsigned tile_d0,
                                                unsigned tile_d1,
                                                unsigned stride0_elems,
                                                unsigned pad_interval_code,
                                                unsigned pad_amount_code) {
  const unsigned long long ga = (unsigned long long)(uintptr_t)gaddr;
  v4u g0;
  g0[0] = 1u;                                            // count=1 (valid, user)
  g0[1] = lds_byte_off;                                  // lds_addr
  g0[2] = (unsigned)(ga & 0xffffffffu);                  // global_addr[31:0]
  g0[3] = (unsigned)((ga >> 32) & 0x01ffffffu) | (2u << 30);  // [56:32] | type=2

  v8i g1;
  g1[0] = (int)((1u << 16) |                             // data_size = 2 bytes
                (1u << 20) |                             // pad_enable
                (pad_interval_code << 22) |
                (pad_amount_code << 25));
  g1[1] = (int)((tensor_d0 & 0xffffu) << 16);            // dim0[15:0] @ bit48
  g1[2] = (int)((tensor_d0 >> 16) | ((tensor_d1 & 0xffffu) << 16));
  g1[3] = (int)((tensor_d1 >> 16) | (tile_d0 << 16));    // tile_dim0 @ bit112
  g1[4] = (int)(tile_d1 & 0xffffu);                      // tile_dim1; tile_dim2=0
  g1[5] = (int)stride0_elems;                            // dim0_stride[31:0]
  g1[6] = 0;                                             // stride[47:32]=0, dim1_stride lo=0
  g1[7] = 0;

  v4i z4;
  z4[0] = 0; z4[1] = 0; z4[2] = 0; z4[3] = 0;
#if __clang_major__ >= 23
  v8i z8;
#pragma unroll
  for (int i = 0; i < 8; ++i) z8[i] = 0;
  __builtin_amdgcn_tensor_load_to_lds(g0, g1, z4, z4, z8, 0);
#else
  __builtin_amdgcn_tensor_load_to_lds(g0, g1, z4, z4, 0);
#endif
}

// ---------------------------------------------------------------------------
// Projection: out[o,l] = sum_c W[o,c] x[b,c,l] + bias[o], cast to f16.
// which (blockIdx.z): 0 -> Q ([L,C]), 1 -> K ([L,C]), 2 -> V ([C,L]).
// ---------------------------------------------------------------------------
__launch_bounds__(128)
__global__ void cgp_proj_kernel(const float* __restrict__ pose,
                                const float* __restrict__ idf,
                                const float* __restrict__ Wq, const float* __restrict__ bq,
                                const float* __restrict__ Wk, const float* __restrict__ bk,
                                const float* __restrict__ Wv, const float* __restrict__ bv,
                                _Float16* __restrict__ Qb,
                                _Float16* __restrict__ Kb,
                                _Float16* __restrict__ Vb) {
  const int which = blockIdx.z;
  const int b     = blockIdx.y;
  const int l0    = blockIdx.x * 32;
  const int o     = threadIdx.x;

  const float* x    = (which == 1) ? idf : pose;
  const float* W    = (which == 0) ? Wq : (which == 1) ? Wk : Wv;
  const float* bias = (which == 0) ? bq : (which == 1) ? bk : bv;

  __shared__ float xs[CC][32];

  const float* xb = x + (size_t)b * CC * LL;
  for (int i = threadIdx.x; i < CC * 32; i += 128) {
    const int c = i >> 5, j = i & 31;
    xs[c][j] = xb[(size_t)c * LL + l0 + j];
  }
  __syncthreads();

  float acc[32];
#pragma unroll
  for (int j = 0; j < 32; ++j) acc[j] = 0.0f;

  for (int c = 0; c < CC; ++c) {
    const float w = W[o * CC + c];
#pragma unroll
    for (int j = 0; j < 32; ++j) acc[j] = fmaf(w, xs[c][j], acc[j]);
  }

  const float bo = bias[o];
  if (which == 2) {
    _Float16* dst = Vb + (size_t)b * CC * LL + (size_t)o * LL + l0;
#pragma unroll
    for (int j = 0; j < 32; ++j) dst[j] = (_Float16)(acc[j] + bo);
  } else {
    _Float16* dst = (which == 0 ? Qb : Kb) + (size_t)b * LL * CC + (size_t)l0 * CC + o;
#pragma unroll
    for (int j = 0; j < 32; ++j) dst[(size_t)j * CC] = (_Float16)(acc[j] + bo);
  }
}

// ---------------------------------------------------------------------------
// Flash attention with TDM double-buffered K/V staging.
// Block = 256 threads = 8 waves; wave w owns query rows [q0+16w, q0+16w+16).
// Wave 0 issues tensor_load_to_lds for tile t+1, waits TENSORcnt<=2 (tile t
// done; TDM ops of one wave complete in order), barrier releases all waves.
// ---------------------------------------------------------------------------
#define KT   64
#define KSTR 136  // K LDS row stride (halves): 64 DW row + 4 DW hw pad
#define VSTR 72   // V LDS row stride (halves): 32 DW row + 4 DW hw pad
#define PSTR 72   // P pad row stride (halves)
#define NIT  (LL / KT)

__launch_bounds__(256)
__global__ void cgp_attn_kernel(const _Float16* __restrict__ Qb,
                                const _Float16* __restrict__ Kb,
                                const _Float16* __restrict__ Vb,
                                const float* __restrict__ pose,
                                const float* __restrict__ gamma,
                                float* __restrict__ out) {
  __shared__ _Float16 sK[2][64 * KSTR];
  __shared__ _Float16 sV[2][CC * VSTR];
  __shared__ _Float16 sP[8 * 16 * PSTR];

  const int b   = blockIdx.y;
  const int q0  = blockIdx.x * 128;
  const int tid = threadIdx.x;
  const int wv  = tid >> 5;
  const int ln  = tid & 31;
  const int hn  = ln & 15;
  const int hg  = ln >> 4;

  const _Float16* Qbase = Qb + (size_t)b * LL * CC;  // [L,C]
  const _Float16* Kbase = Kb + (size_t)b * LL * CC;  // [L,C]
  const _Float16* Vbase = Vb + (size_t)b * CC * LL;  // [C,L]

  const int i0 = q0 + wv * 16;
  const bool tdm_wave = (wv == 0);

  // Resident Q fragments.
  v16h qf[4];
#pragma unroll
  for (int kc = 0; kc < 4; ++kc)
    qf[kc] = frag_rowmajor(Qbase, i0 + hn, CC, kc * 32 + hg * 8);

  v8f O[8];
#pragma unroll
  for (int cc8 = 0; cc8 < 8; ++cc8) O[cc8] = zero_v8f();

  float mrun[8], lrun[8];
#pragma unroll
  for (int r = 0; r < 8; ++r) { mrun[r] = -3.0e38f; lrun[r] = 0.0f; }

  _Float16* myP = &sP[wv * 16 * PSTR];

  // Prologue: DMA tile 0 into buffer 0.
  if (tdm_wave) {
    tdm_load_2d_f16((unsigned)(uintptr_t)&sK[0][0], Kbase,
                    CC, LL, CC, KT, CC, /*pad: every 64 DW*/ 5, /*4 DW*/ 3);
    tdm_load_2d_f16((unsigned)(uintptr_t)&sV[0][0], Vbase,
                    LL, CC, KT, CC, LL, /*pad: every 32 DW*/ 4, /*4 DW*/ 3);
  }

  for (int it = 0; it < NIT; ++it) {
    const int nb = (it + 1) & 1;
    if (tdm_wave) {
      if (it + 1 < NIT) {
        const int j1 = (it + 1) * KT;
        tdm_load_2d_f16((unsigned)(uintptr_t)&sK[nb][0],
                        Kbase + (size_t)j1 * CC, CC, LL, CC, KT, CC, 5, 3);
        tdm_load_2d_f16((unsigned)(uintptr_t)&sV[nb][0],
                        Vbase + j1, LL, CC, KT, CC, LL, 4, 3);
        __builtin_amdgcn_s_wait_tensorcnt(2);   // current tile's 2 ops done
      } else {
        __builtin_amdgcn_s_wait_tensorcnt(0);
      }
    }
    __syncthreads();  // tile `it` visible to all waves

    const _Float16* k = sK[it & 1];
    const _Float16* v = sV[it & 1];

    // ---- S = Q K^T : preload the 4 K-fragments, then the 4-WMMA chain ----
    v8f S[4];
#pragma unroll
    for (int jt = 0; jt < 4; ++jt) {
      v16h kf[4];
#pragma unroll
      for (int kc = 0; kc < 4; ++kc)
        kf[kc] = frag_rowmajor(k, jt * 16 + hn, KSTR, kc * 32 + hg * 8);
      v8f acc = zero_v8f();
#pragma unroll
      for (int kc = 0; kc < 4; ++kc)
        acc = __builtin_amdgcn_wmma_f32_16x16x32_f16(false, qf[kc], false, kf[kc],
                                                     (short)0, acc, false, false);
      S[jt] = acc;
    }

    // ---- online softmax (rows m = hg*8 + r live in one VGPR per half) ----
    float mnew[8], alpha[8];
#pragma unroll
    for (int r = 0; r < 8; ++r) {
      float mv = fmaxf(fmaxf(S[0][r], S[1][r]), fmaxf(S[2][r], S[3][r]));
#pragma unroll
      for (int off = 8; off >= 1; off >>= 1) mv = fmaxf(mv, __shfl_xor(mv, off, 32));
      mnew[r]  = fmaxf(mrun[r], mv);
      alpha[r] = __expf(mrun[r] - mnew[r]);
      mrun[r]  = mnew[r];
    }
#pragma unroll
    for (int jt = 0; jt < 4; ++jt)
#pragma unroll
      for (int r = 0; r < 8; ++r) S[jt][r] = __expf(S[jt][r] - mnew[r]);
#pragma unroll
    for (int r = 0; r < 8; ++r) {
      float s = (S[0][r] + S[1][r]) + (S[2][r] + S[3][r]);
#pragma unroll
      for (int off = 8; off >= 1; off >>= 1) s += __shfl_xor(s, off, 32);
      lrun[r] = lrun[r] * alpha[r] + s;
    }
#pragma unroll
    for (int cc8 = 0; cc8 < 8; ++cc8)
#pragma unroll
      for (int r = 0; r < 8; ++r) O[cc8][r] *= alpha[r];

    // ---- P: D-layout -> f16 A-layout via per-wave LDS pad (in-order LDS) ----
#pragma unroll
    for (int jt = 0; jt < 4; ++jt)
#pragma unroll
      for (int r = 0; r < 8; ++r)
        myP[(hg * 8 + r) * PSTR + jt * 16 + hn] = (_Float16)S[jt][r];

    // ---- O += P V^T : 2 j-chunks x 8 independent accumulation chains ----
#pragma unroll
    for (int jc = 0; jc < 2; ++jc) {
      v16h pf = frag_rowmajor(myP, hn, PSTR, jc * 32 + hg * 8);
#pragma unroll
      for (int cc8 = 0; cc8 < 8; ++cc8) {
        v16h vf = frag_rowmajor(v, cc8 * 16 + hn, VSTR, jc * 32 + hg * 8);
        O[cc8] = __builtin_amdgcn_wmma_f32_16x16x32_f16(false, pf, false, vf,
                                                        (short)0, O[cc8], false, false);
      }
    }
    __syncthreads();  // all waves done with buf[it&1] before TDM reuses it
  }

  // ---- epilogue: out = pose + gamma * O / l ----
  float inv[8];
#pragma unroll
  for (int r = 0; r < 8; ++r) inv[r] = 1.0f / lrun[r];
  const float g = gamma[0];
  const float* pbase = pose + (size_t)b * CC * LL;
  float*       obase = out  + (size_t)b * CC * LL;

#pragma unroll
  for (int cc8 = 0; cc8 < 8; ++cc8) {
    const size_t base = (size_t)(cc8 * 16 + hn) * LL + (size_t)(i0 + hg * 8);
    const float4 p0 = *reinterpret_cast<const float4*>(pbase + base);
    const float4 p1 = *reinterpret_cast<const float4*>(pbase + base + 4);
    float4 r0, r1;
    r0.x = fmaf(g, O[cc8][0] * inv[0], p0.x);
    r0.y = fmaf(g, O[cc8][1] * inv[1], p0.y);
    r0.z = fmaf(g, O[cc8][2] * inv[2], p0.z);
    r0.w = fmaf(g, O[cc8][3] * inv[3], p0.w);
    r1.x = fmaf(g, O[cc8][4] * inv[4], p1.x);
    r1.y = fmaf(g, O[cc8][5] * inv[5], p1.y);
    r1.z = fmaf(g, O[cc8][6] * inv[6], p1.z);
    r1.w = fmaf(g, O[cc8][7] * inv[7], p1.w);
    *reinterpret_cast<float4*>(obase + base)     = r0;
    *reinterpret_cast<float4*>(obase + base + 4) = r1;
  }
}

// ---------------------------------------------------------------------------
extern "C" void kernel_launch(void* const* d_in, const int* in_sizes, int n_in,
                              void* d_out, int out_size, void* d_ws, size_t ws_size,
                              hipStream_t stream) {
  const float* pose  = (const float*)d_in[0];
  const float* idf   = (const float*)d_in[1];
  const float* Wq    = (const float*)d_in[2];
  const float* bq    = (const float*)d_in[3];
  const float* Wk    = (const float*)d_in[4];
  const float* bk    = (const float*)d_in[5];
  const float* Wv    = (const float*)d_in[6];
  const float* bv    = (const float*)d_in[7];
  const float* gamma = (const float*)d_in[8];
  float* out = (float*)d_out;

  const size_t mat_bytes = (size_t)BB * LL * CC * sizeof(_Float16);  // 8 MB each
  char* ws = (char*)d_ws;
  _Float16* Qb = reinterpret_cast<_Float16*>(ws);
  _Float16* Kb = reinterpret_cast<_Float16*>(ws + mat_bytes);
  _Float16* Vb = reinterpret_cast<_Float16*>(ws + 2 * mat_bytes);

  cgp_proj_kernel<<<dim3(LL / 32, BB, 3), dim3(128), 0, stream>>>(
      pose, idf, Wq, bq, Wk, bk, Wv, bv, Qb, Kb, Vb);

  cgp_attn_kernel<<<dim3(LL / 128, BB), dim3(256), 0, stream>>>(
      Qb, Kb, Vb, pose, gamma, out);
}
